// PreferenceConditionedDecoder_10033043603494
// MI455X (gfx1250) — matrix operand order repaired and checked by reading, not compiled, not run on hardware
//
#include <hip/hip_runtime.h>
#include <hip/hip_bf16.h>
#include <math.h>

// ---------------------------------------------------------------------------
// PreferenceConditionedDecoder for MI455X (gfx1250, wave32, WMMA)
//   H=256, NH=8, E=512, Dh=64, B=32, T=2048, R=16, steps=32
// Strategy:
//   * K/V projection: bf16 WMMA GEMM (65536x512x512), K/V stored bf16 so that
//     K+V (134MB) fits in the 192MB L2 across all 32 attention sweeps.
//   * Weights pre-packed once into WMMA B-fragment order (bf16) -> single
//     aligned 32B load per lane per fragment.
//   * All step activations (h/ctx/pref/attn/hid) kept in bf16 so per-step
//     GEMM A-fragments are two aligned 16B loads (no cvt VALU in hot loop).
//   * Per-step GEMMs: 2 n-tiles per wave (A reuse + 2 independent WMMA chains).
//   * Attention: fused scores+softmax+A.V per (b,h) block, LDS-resident w[T].
// ---------------------------------------------------------------------------

#define DEVFN static __device__ __forceinline__

typedef __attribute__((ext_vector_type(16))) __bf16        bf16x16;
typedef __attribute__((ext_vector_type(8)))  unsigned int  u32x8;
typedef __attribute__((ext_vector_type(8)))  float         f32x8;

DEVFN unsigned f2bfbits(float f) {
  unsigned u = __builtin_bit_cast(unsigned, f);
  return (u + 0x7FFFu + ((u >> 16) & 1u)) >> 16;   // RNE fp32 -> bf16
}
DEVFN unsigned pack2bf(float lo, float hi) {
  return f2bfbits(lo) | (f2bfbits(hi) << 16);
}
DEVFN float bf2f(unsigned short v) {
  return __builtin_bit_cast(float, ((unsigned)v) << 16);
}
DEVFN float sigmoidf_(float x) { return 1.0f / (1.0f + __expf(-x)); }

DEVFN f32x8 wmma_bf16(u32x8 a, u32x8 b, f32x8 c) {
  return __builtin_amdgcn_wmma_f32_16x16x32_bf16(
      false, __builtin_bit_cast(bf16x16, a),
      false, __builtin_bit_cast(bf16x16, b),
      (short)0, c, false, false);
}

// A-fragment (16x32 bf16) from fp32 row-major A (cvt in registers).
// ISA 16-bit A layout: lanes 0-15: M=lane, VGPR j holds K pair
// {j/4*16 + (j%4)*2, +1}; lanes 16-31 same rows, K offset +8.
DEVFN u32x8 load_a_frag_f32(const float* __restrict__ A, int lda, int mBase, int kBase) {
  int lane = threadIdx.x & 31;
  const float* ap = A + (size_t)(mBase + (lane & 15)) * lda + kBase + (lane >> 4) * 8;
  u32x8 r;
#pragma unroll
  for (int j = 0; j < 8; ++j) {
    int k = (j >> 2) * 16 + (j & 3) * 2;
    float2 f = *(const float2*)(ap + k);
    r[j] = pack2bf(f.x, f.y);
  }
  return r;
}

// A-fragment straight from bf16 row-major A: two aligned 16B loads per lane.
DEVFN u32x8 load_a_frag_bf16(const unsigned short* __restrict__ A, int lda,
                             int mBase, int kBase) {
  int lane = threadIdx.x & 31;
  const unsigned short* ap =
      A + (size_t)(mBase + (lane & 15)) * lda + kBase + (lane >> 4) * 8;
  union { u32x8 v; uint4 q[2]; } u;
  u.q[0] = *(const uint4*)(ap);        // j0..3: K 0..7
  u.q[1] = *(const uint4*)(ap + 16);   // j4..7: K 16..23
  return u.v;
}

// B-fragment (32x16 bf16) from pre-packed weights: one aligned 32B load/lane.
DEVFN u32x8 load_b_frag_packed(const unsigned* __restrict__ Bp, int nTiles, int kt, int nt) {
  int lane = threadIdx.x & 31;
  const unsigned* p = Bp + ((size_t)(kt * nTiles + nt) * 32 + lane) * 8;
  return *(const u32x8*)p;
}

// ---------------------------------------------------------------------------
// One-time weight packing: B[k][n] = src[k*kStride + n*nStride] -> fragment
// order. Output dword idx = ((kt*nTiles + nt)*32 + lane)*8 + j,
// lane: n = nt*16 + lane%16 ; k = kt*32 + (lane/16)*8 + (j/4)*16 + (j%4)*2 (pair k,k+1).
// ---------------------------------------------------------------------------
__global__ void pack_b_kernel(const float* __restrict__ src, int kStride, int nStride,
                              int K, int N, unsigned* __restrict__ dst) {
  int idx = blockIdx.x * blockDim.x + threadIdx.x;
  int total = (K * N) >> 1;
  if (idx >= total) return;
  int j    = idx & 7;
  int lane = (idx >> 3) & 31;
  int tile = idx >> 8;
  int nTiles = N >> 4;
  int kt = tile / nTiles;
  int nt = tile - kt * nTiles;
  int n = nt * 16 + (lane & 15);
  int k = kt * 32 + (lane >> 4) * 8 + (j >> 2) * 16 + (j & 3) * 2;
  float f0 = src[(size_t)k * kStride + (size_t)n * nStride];
  float f1 = src[(size_t)(k + 1) * kStride + (size_t)n * nStride];
  dst[idx] = pack2bf(f0, f1);
}

__global__ void bias_sum_kernel(const float* a, const float* b, float* out, int n) {
  int i = blockIdx.x * blockDim.x + threadIdx.x;
  if (i < n) out[i] = a[i] + b[i];
}

// ---------------------------------------------------------------------------
// K/V projection: C(65536x512) = A(fp32) x Bp(packed bf16) + bias, bf16 out.
// Block = 8 waves -> 32x512 tile; wave = 16 rows x 128 cols (8 accumulators).
// ---------------------------------------------------------------------------
__global__ __launch_bounds__(256) void kv_proj_kernel(
    const float* __restrict__ A, const unsigned* __restrict__ Bp,
    const float* __restrict__ bias, unsigned short* __restrict__ out) {
  const int E = 512;
  int wave = threadIdx.x >> 5;
  int lane = threadIdx.x & 31;
  int mBase = blockIdx.x * 32 + (wave & 1) * 16;
  int ntBase = (wave >> 1) * 8;           // 8 n-tiles per wave (128 cols)
  f32x8 acc[8] = {};
  const int nTiles = E >> 4;              // 32
#pragma unroll 1
  for (int kt = 0; kt < 16; ++kt) {
    u32x8 a = load_a_frag_f32(A, E, mBase, kt * 32);
    if (kt + 1 < 16)                       // keep next A strip moving (L2)
      __builtin_prefetch(A + (size_t)(mBase + (lane & 15)) * E + (kt + 1) * 32, 0, 1);
#pragma unroll
    for (int t = 0; t < 8; ++t) {
      u32x8 b = load_b_frag_packed(Bp, nTiles, kt, ntBase + t);
      acc[t] = wmma_bf16(a, b, acc[t]);
    }
  }
  int col = lane & 15;
  int rOff = (lane >> 4) * 8;              // C layout: lanes16-31 -> M+8
#pragma unroll
  for (int t = 0; t < 8; ++t) {
    int n = (ntBase + t) * 16 + col;
    float bb = bias[n];
#pragma unroll
    for (int r = 0; r < 8; ++r) {
      int row = mBase + rOff + r;
      out[(size_t)row * E + n] = (unsigned short)f2bfbits(acc[t][r] + bb);
    }
  }
}

// ---------------------------------------------------------------------------
// Generic small GEMM, bf16 A (row-major) x packed-bf16 B.
//   C = (A x B + bias) * scale ; flags: bit0 = relu, bit1 = bf16 output.
// Each wave owns 2 adjacent n-tiles (A reuse, 2 independent WMMA chains).
// ---------------------------------------------------------------------------
DEVFN void gemm_epilogue(f32x8 acc, const float* bias, void* C, int ldc,
                         int mt, int nt, float scale, int flags) {
  int lane = threadIdx.x & 31;
  int n = nt * 16 + (lane & 15);
  int rOff = (lane >> 4) * 8;
  float bb = bias ? bias[n] : 0.0f;
#pragma unroll
  for (int r = 0; r < 8; ++r) {
    float v = (acc[r] + bb) * scale;
    if (flags & 1) v = fmaxf(v, 0.0f);
    size_t idx = (size_t)(mt * 16 + rOff + r) * ldc + n;
    if (flags & 2) ((unsigned short*)C)[idx] = (unsigned short)f2bfbits(v);
    else           ((float*)C)[idx] = v;
  }
}

__global__ __launch_bounds__(256) void gemm_wmma_kernel(
    const unsigned short* __restrict__ A, int lda, const unsigned* __restrict__ Bp,
    const float* __restrict__ bias, void* __restrict__ C, int ldc,
    int M, int N, int K, float scale, int flags) {
  int wglob = (blockIdx.x * blockDim.x + threadIdx.x) >> 5;
  int gridWaves = (gridDim.x * blockDim.x) >> 5;
  int mTiles = M >> 4, nTiles = N >> 4;
  int nPairs = (nTiles + 1) >> 1;
  int total = mTiles * nPairs;
  for (int pair = wglob; pair < total; pair += gridWaves) {
    int mt = pair % mTiles;
    int nt0 = (pair / mTiles) * 2;
    bool two = (nt0 + 1) < nTiles;
    f32x8 acc0 = {}, acc1 = {};
    for (int kt = 0; kt < (K >> 5); ++kt) {
      u32x8 a = load_a_frag_bf16(A, lda, mt * 16, kt * 32);
      u32x8 b0 = load_b_frag_packed(Bp, nTiles, kt, nt0);
      acc0 = wmma_bf16(a, b0, acc0);
      if (two) {
        u32x8 b1 = load_b_frag_packed(Bp, nTiles, kt, nt0 + 1);
        acc1 = wmma_bf16(a, b1, acc1);
      }
    }
    gemm_epilogue(acc0, bias, C, ldc, mt, nt0, scale, flags);
    if (two) gemm_epilogue(acc1, bias, C, ldc, mt, nt0 + 1, scale, flags);
  }
}

// ---------------------------------------------------------------------------
// Fused attention per (b,h): scores over T=2048 (bf16 K), softmax, A.V.
// qh fp32 in, attn bf16 out (feeds the Wo GEMM as A).
// ---------------------------------------------------------------------------
__global__ __launch_bounds__(256) void attn_kernel(
    const float* __restrict__ qh, const unsigned short* __restrict__ Kbf,
    const unsigned short* __restrict__ Vbf, unsigned short* __restrict__ attn) {
  const int T = 2048, E = 512;
  int b = blockIdx.x >> 3;
  int h = blockIdx.x & 7;
  __shared__ float q_s[64];
  __shared__ float w_s[2048];
  __shared__ float red[256];
  int tid = threadIdx.x;
  if (tid < 64) q_s[tid] = qh[b * E + h * 64 + tid];
  __syncthreads();

  const unsigned short* Kb = Kbf + (size_t)b * T * E + h * 64;
  float sloc[8];
  float mloc = -1e30f;
#pragma unroll
  for (int i = 0; i < 8; ++i) {
    int t = tid + i * 256;
    const uint4* kr = (const uint4*)(Kb + (size_t)t * E);
    float s = 0.0f;
#pragma unroll
    for (int c = 0; c < 8; ++c) {
      uint4 u = kr[c];
      int d0 = c * 8;
      s += q_s[d0 + 0] * bf2f((unsigned short)u.x) + q_s[d0 + 1] * bf2f((unsigned short)(u.x >> 16));
      s += q_s[d0 + 2] * bf2f((unsigned short)u.y) + q_s[d0 + 3] * bf2f((unsigned short)(u.y >> 16));
      s += q_s[d0 + 4] * bf2f((unsigned short)u.z) + q_s[d0 + 5] * bf2f((unsigned short)(u.z >> 16));
      s += q_s[d0 + 6] * bf2f((unsigned short)u.w) + q_s[d0 + 7] * bf2f((unsigned short)(u.w >> 16));
    }
    sloc[i] = s;
    mloc = fmaxf(mloc, s);
  }
  red[tid] = mloc; __syncthreads();
  for (int off = 128; off > 0; off >>= 1) {
    if (tid < off) red[tid] = fmaxf(red[tid], red[tid + off]);
    __syncthreads();
  }
  float m = red[0]; __syncthreads();
  float ssum = 0.0f;
#pragma unroll
  for (int i = 0; i < 8; ++i) {
    float e = __expf(sloc[i] - m);
    w_s[tid + i * 256] = e;
    ssum += e;
  }
  red[tid] = ssum; __syncthreads();
  for (int off = 128; off > 0; off >>= 1) {
    if (tid < off) red[tid] += red[tid + off];
    __syncthreads();
  }
  float inv = 1.0f / red[0]; __syncthreads();

  // A.V: d-parallel (coalesced across lanes), 4 t-groups reduced in LDS.
  int d = tid & 63, grp = tid >> 6;
  const unsigned short* Vb = Vbf + (size_t)b * T * E + h * 64 + d;
  float acc = 0.0f;
  for (int t = grp; t < T; t += 4) acc += w_s[t] * bf2f(Vb[(size_t)t * E]);
  red[tid] = acc; __syncthreads();
  if (grp == 0) {
    float v = (red[d] + red[d + 64] + red[d + 128] + red[d + 192]) * inv;
    attn[b * E + h * 64 + d] = (unsigned short)f2bfbits(v);
  }
}

// ---------------------------------------------------------------------------
// Misc small kernels (activation buffers are bf16 = ushort)
// ---------------------------------------------------------------------------
__global__ void init_pref_kernel(const float* __restrict__ pref, const float* __restrict__ W,
                                 const float* __restrict__ bias, const float* __restrict__ g,
                                 const float* __restrict__ b2, unsigned short* qbuf,
                                 unsigned short* xh, unsigned short* dec) {
  int m = blockIdx.x, j = threadIdx.x;  // 32 blocks x 256
  __shared__ float red[256];
  float p0 = pref[m * 2], p1 = pref[m * 2 + 1];
  float t = fmaxf(p0 * W[j] + p1 * W[256 + j] + bias[j], 0.0f);
  red[j] = t; __syncthreads();
  for (int off = 128; off > 0; off >>= 1) { if (j < off) red[j] += red[j + off]; __syncthreads(); }
  float mean = red[0] * (1.0f / 256.0f); __syncthreads();
  float dcen = t - mean;
  red[j] = dcen * dcen; __syncthreads();
  for (int off = 128; off > 0; off >>= 1) { if (j < off) red[j] += red[j + off]; __syncthreads(); }
  float var = red[0] * (1.0f / 256.0f);
  float y = dcen * rsqrtf(var + 1e-5f) * g[j] + b2[j];
  unsigned short yb = (unsigned short)f2bfbits(y);
  qbuf[m * 512 + 256 + j] = yb;     // q = [h, pref]
  xh[m * 1024 + 512 + j] = yb;      // xh = [h, ctx, pref, h]
  dec[m * 768 + 512 + j] = yb;      // dec = [h_new, ctx, pref]
}

__global__ void init_state_kernel(const float* __restrict__ context, unsigned short* qbuf,
                                  unsigned short* xh, float* cbuf) {
  int idx = blockIdx.x * blockDim.x + threadIdx.x;  // 8192
  int m = idx >> 8, n = idx & 255;
  unsigned short h0 = (unsigned short)f2bfbits(context[idx]);
  qbuf[m * 512 + n] = h0;
  xh[m * 1024 + n] = h0;
  xh[m * 1024 + 768 + n] = h0;
  cbuf[idx] = 0.0f;
}

__global__ void copy_ctx_kernel(const float* __restrict__ ctx, unsigned short* xh,
                                unsigned short* dec) {
  int idx = blockIdx.x * blockDim.x + threadIdx.x;  // 8192 (first H cols of ctx)
  int m = idx >> 8, n = idx & 255;
  unsigned short v = (unsigned short)f2bfbits(ctx[m * 512 + n]);
  xh[m * 1024 + 256 + n] = v;
  dec[m * 768 + 256 + n] = v;
}

__global__ void lstm_kernel(const float* __restrict__ gates, float* cbuf,
                            unsigned short* qbuf, unsigned short* xh, unsigned short* dec) {
  int idx = blockIdx.x * blockDim.x + threadIdx.x;  // 8192
  int m = idx >> 8, n = idx & 255;
  const float* g = gates + m * 1024;
  float i_ = sigmoidf_(g[n]);
  float f_ = sigmoidf_(g[256 + n]);
  float gg = tanhf(g[512 + n]);
  float o_ = sigmoidf_(g[768 + n]);
  float c = f_ * cbuf[idx] + i_ * gg;
  float h = o_ * tanhf(c);
  cbuf[idx] = c;
  unsigned short hb = (unsigned short)f2bfbits(h);
  qbuf[m * 512 + n] = hb;
  xh[m * 1024 + n] = hb;
  xh[m * 1024 + 768 + n] = hb;
  dec[m * 768 + n] = hb;
}

// ---------------------------------------------------------------------------
extern "C" void kernel_launch(void* const* d_in, const int* in_sizes, int n_in,
                              void* d_out, int out_size, void* d_ws, size_t ws_size,
                              hipStream_t stream) {
  (void)in_sizes; (void)n_in; (void)out_size; (void)ws_size;
  const int B = 32, T = 2048, E = 512, NSTEPS = 32;
  const float scaleQ = 0.125f;  // 1/sqrt(64)

  const float* enc   = (const float*)d_in[0];
  const float* ctxin = (const float*)d_in[1];
  const float* prefv = (const float*)d_in[2];
  const float* prefW = (const float*)d_in[3];
  const float* prefB = (const float*)d_in[4];
  const float* lnG   = (const float*)d_in[5];
  const float* lnB   = (const float*)d_in[6];
  const float* Wq = (const float*)d_in[7];  const float* bq = (const float*)d_in[8];
  const float* Wk = (const float*)d_in[9];  const float* bk = (const float*)d_in[10];
  const float* Wv = (const float*)d_in[11]; const float* bv = (const float*)d_in[12];
  const float* Wo = (const float*)d_in[13]; const float* bo = (const float*)d_in[14];
  const float* Wih = (const float*)d_in[15]; const float* bih = (const float*)d_in[16];
  const float* Whh = (const float*)d_in[17]; const float* bhh = (const float*)d_in[18];
  const float* hW1 = (const float*)d_in[19]; const float* hb1 = (const float*)d_in[20];
  const float* hW2 = (const float*)d_in[21]; const float* hb2 = (const float*)d_in[22];
  float* out = (float*)d_out;

  // Workspace carve (256B aligned)
  size_t off = 0;
  auto carve = [&](size_t bytes) -> void* {
    off = (off + 255) & ~(size_t)255;
    void* p = (char*)d_ws + off;
    off += bytes;
    return p;
  };
  unsigned short* Kbf = (unsigned short*)carve((size_t)B * T * E * 2);
  unsigned short* Vbf = (unsigned short*)carve((size_t)B * T * E * 2);
  unsigned* WqP  = (unsigned*)carve(512 * 512 * 2);
  unsigned* WkP  = (unsigned*)carve(512 * 512 * 2);
  unsigned* WvP  = (unsigned*)carve(512 * 512 * 2);
  unsigned* WoP  = (unsigned*)carve(512 * 512 * 2);
  unsigned* WgP  = (unsigned*)carve(1024 * 1024 * 2);  // [W_ih^T ; W_hh^T]
  unsigned* h1P  = (unsigned*)carve(768 * 256 * 2);
  unsigned* h2P  = (unsigned*)carve(256 * 16 * 2);
  unsigned short* qbuf  = (unsigned short*)carve(B * 512 * 2);   // [h, pref] bf16
  unsigned short* xh    = (unsigned short*)carve(B * 1024 * 2);  // [h, ctx, pref, h] bf16
  unsigned short* dec   = (unsigned short*)carve(B * 768 * 2);   // [h, ctx, pref] bf16
  unsigned short* attnb = (unsigned short*)carve(B * 512 * 2);   // attn bf16
  unsigned short* hid   = (unsigned short*)carve(B * 256 * 2);   // relu hidden bf16
  float* qh    = (float*)carve(B * 512 * 4);
  float* ctxb  = (float*)carve(B * 512 * 4);
  float* gates = (float*)carve(B * 1024 * 4);
  float* cbuf  = (float*)carve(B * 256 * 4);
  float* biasg = (float*)carve(1024 * 4);

  // ---- one-time: pack weights into WMMA fragment order (bf16) ----
  auto packGrid = [](int K, int N) { return (K * N / 2 + 255) / 256; };
  pack_b_kernel<<<packGrid(512, 512), 256, 0, stream>>>(Wq, 512, 1, 512, 512, WqP);
  pack_b_kernel<<<packGrid(512, 512), 256, 0, stream>>>(Wk, 512, 1, 512, 512, WkP);
  pack_b_kernel<<<packGrid(512, 512), 256, 0, stream>>>(Wv, 512, 1, 512, 512, WvP);
  pack_b_kernel<<<packGrid(512, 512), 256, 0, stream>>>(Wo, 512, 1, 512, 512, WoP);
  // gates B = [W_ih^T (768x1024) ; W_hh^T (256x1024)]  (transposed sources)
  pack_b_kernel<<<packGrid(768, 1024), 256, 0, stream>>>(Wih, 1, 768, 768, 1024, WgP);
  pack_b_kernel<<<packGrid(256, 1024), 256, 0, stream>>>(Whh, 1, 256, 256, 1024,
                                                         WgP + (768 * 1024 / 2));
  pack_b_kernel<<<packGrid(768, 256), 256, 0, stream>>>(hW1, 256, 1, 768, 256, h1P);
  pack_b_kernel<<<packGrid(256, 16), 256, 0, stream>>>(hW2, 16, 1, 256, 16, h2P);
  bias_sum_kernel<<<4, 256, 0, stream>>>(bih, bhh, biasg, 1024);

  // ---- one-time: state init + pref LN ----
  init_state_kernel<<<32, 256, 0, stream>>>(ctxin, qbuf, xh, cbuf);
  init_pref_kernel<<<32, 256, 0, stream>>>(prefv, prefW, prefB, lnG, lnB, qbuf, xh, dec);

  // ---- one-time: K/V projections (bf16 WMMA, bf16 out -> L2-resident) ----
  kv_proj_kernel<<<(B * T) / 32, 256, 0, stream>>>(enc, WkP, bk, Kbf);
  kv_proj_kernel<<<(B * T) / 32, 256, 0, stream>>>(enc, WvP, bv, Vbf);

  // ---- decode loop ----
  auto gemmGrid = [](int M, int N) {
    int pairs = (M / 16) * (((N / 16) + 1) / 2);
    return (pairs + 7) / 8;
  };
  for (int s = 0; s < NSTEPS; ++s) {
    // qh = ([h, pref] @ Wq + bq) * scale
    gemm_wmma_kernel<<<gemmGrid(32, 512), 256, 0, stream>>>(
        qbuf, 512, WqP, bq, qh, 512, 32, 512, 512, scaleQ, 0);
    // attention (fused scores/softmax/AV), 256 (b,h) blocks
    attn_kernel<<<B * 8, 256, 0, stream>>>(qh, Kbf, Vbf, attnb);
    // ctx = attn @ Wo + bo
    gemm_wmma_kernel<<<gemmGrid(32, 512), 256, 0, stream>>>(
        attnb, 512, WoP, bo, ctxb, 512, 32, 512, 512, 1.0f, 0);
    copy_ctx_kernel<<<32, 256, 0, stream>>>(ctxb, xh, dec);
    // gates = [x, h] @ [W_ih^T; W_hh^T] + (b_ih + b_hh)
    gemm_wmma_kernel<<<gemmGrid(32, 1024), 256, 0, stream>>>(
        xh, 1024, WgP, biasg, gates, 1024, 32, 1024, 1024, 1.0f, 0);
    lstm_kernel<<<32, 256, 0, stream>>>(gates, cbuf, qbuf, xh, dec);
    // hid = relu(dec @ hW1 + hb1), bf16 out (feeds logits GEMM)
    gemm_wmma_kernel<<<gemmGrid(32, 256), 256, 0, stream>>>(
        dec, 768, h1P, hb1, hid, 256, 32, 256, 768, 1.0f, 1 | 2);
    // logits -> out[b, s, r]  (row stride 32*16 = 512, col offset s*16)
    gemm_wmma_kernel<<<1, 256, 0, stream>>>(
        hid, 256, h2P, hb2, out + s * 16, NSTEPS * 16, 32, 16, 256, 1.0f, 0);
  }
}
